// NeuroBranch_14302241096156
// MI455X (gfx1250) — matrix operand order repaired and checked by reading, not compiled, not run on hardware
//
#include <hip/hip_runtime.h>
#include <math.h>

#define DHID 128
#define WPB 4            // waves per block
#define RPB 64           // rows per block (4 waves * 16)

typedef __attribute__((ext_vector_type(16))) __bf16 v16bf;
typedef __attribute__((ext_vector_type(8)))  float  v8f;

static __device__ __forceinline__ v8f wmma_bf16(v16bf a, v16bf b, v8f c) {
  // D = A(16x32 bf16) * B(32x16 bf16) + C(16x16 f32)
  return __builtin_amdgcn_wmma_f32_16x16x32_bf16(false, a, false, b, (short)0, c, false, false);
}

// A-fragment (16x32 bf16) from a global fp32 row, CDNA5 layout:
// lane: m = lane&15 (caller picks row ptr), kbase = lane<16 ? 0 : 8
// elem i<8: k = kbase+i ; elem i>=8: k = kbase+16+(i-8)
static __device__ __forceinline__ v16bf load_a_global(const float* rowPtr, int colBase,
                                                      int kbA, float scale) {
  v16bf a;
#pragma unroll
  for (int i = 0; i < 8; ++i)
    a[i] = (__bf16)(rowPtr[colBase + kbA + i] * scale);
#pragma unroll
  for (int i = 0; i < 8; ++i)
    a[8 + i] = (__bf16)(rowPtr[colBase + kbA + 16 + i] * scale);
  return a;
}

static __device__ __forceinline__ v16bf load_a_lds(const __bf16* rowPtr, int colBase, int kbA) {
  v16bf a;
#pragma unroll
  for (int i = 0; i < 8; ++i) a[i] = rowPtr[colBase + kbA + i];
#pragma unroll
  for (int i = 0; i < 8; ++i) a[8 + i] = rowPtr[colBase + kbA + 16 + i];
  return a;
}

// B fragment: pre-packed so each lane reads 16 contiguous bf16 (32 bytes)
static __device__ __forceinline__ v16bf load_b(const __bf16* Wp, int kt, int nt, int lane) {
  return *(const v16bf*)(Wp + ((((size_t)kt * 8 + nt) * 32 + (size_t)lane) << 4));
}

// Pack W[K][128] (fp32, row-major) into per-fragment bf16 B-layout.
// B layout (32x16 bf16): n = nt*16 + (lane&15); k = kt*32 + (lane<16?0:8) + (i<8 ? i : i+8) (i=0..15)
__global__ void pack_kernel(const float* __restrict__ W, __bf16* __restrict__ out, int KT) {
  int frag = blockIdx.x * (blockDim.x >> 5) + (threadIdx.x >> 5);
  int lane = threadIdx.x & 31;
  if (frag >= KT * 8) return;
  int kt = frag >> 3, nt = frag & 7;
  int n  = nt * 16 + (lane & 15);
  int kb = kt * 32 + ((lane < 16) ? 0 : 8);
  __bf16* o = out + (((size_t)frag * 32 + (size_t)lane) << 4);
#pragma unroll
  for (int i = 0; i < 16; ++i)
    o[i] = (__bf16)W[(size_t)(kb + (i < 8 ? i : i + 8)) * DHID + n];
}

__global__ void fill_kernel(float* __restrict__ dst, const float* __restrict__ val, long n) {
  float v = *val;
  long i = blockIdx.x * (long)blockDim.x + threadIdx.x;
  long stride = (long)gridDim.x * blockDim.x;
  for (; i < n; i += stride) dst[i] = v;
}

// msgs[dIdx[e]][:] += src[sIdx[e]][:]   (one edge x 4-float chunk per thread)
__global__ void scatter_kernel(const float* __restrict__ src, float* dst,
                               const int* __restrict__ sIdx, const int* __restrict__ dIdx,
                               int E) {
  long tid = blockIdx.x * (long)blockDim.x + threadIdx.x;
  if (tid >= (long)E * 32) return;
  int e = (int)(tid >> 5);
  int q = ((int)tid & 31) * 4;
  const float4 v = *(const float4*)(src + (size_t)sIdx[e] * DHID + q);
  float* d = dst + (size_t)dIdx[e] * DHID + q;
  __hip_atomic_fetch_add(d + 0, v.x, __ATOMIC_RELAXED, __HIP_MEMORY_SCOPE_AGENT);
  __hip_atomic_fetch_add(d + 1, v.y, __ATOMIC_RELAXED, __HIP_MEMORY_SCOPE_AGENT);
  __hip_atomic_fetch_add(d + 2, v.z, __ATOMIC_RELAXED, __HIP_MEMORY_SCOPE_AGENT);
  __hip_atomic_fetch_add(d + 3, v.w, __ATOMIC_RELAXED, __HIP_MEMORY_SCOPE_AGENT);
}

// Fused 2-layer MLP + row-normalize(ddof=1) + residual.
// X = concat(S[m], msgs[m]*msgScale [, S[flip(m)] if FLIP])  -> relu(X@W1+b1)@W2+b2
// out[m] = normalize(row) + resid[m]
template <bool FLIP>
__global__ void __launch_bounds__(128)
mlp_round_kernel(const float* __restrict__ S, const float* __restrict__ msgs,
                 const float* __restrict__ msgScale,
                 const __bf16* __restrict__ W1p, const float* __restrict__ b1,
                 const __bf16* __restrict__ W2p, const float* __restrict__ b2,
                 const float* resid, float* out, int M, int nv) {
  __shared__ __bf16 Hb[WPB][16 * DHID];
  __shared__ float  Of[WPB][16 * DHID];
  const int wave = threadIdx.x >> 5;
  const int lane = threadIdx.x & 31;
  const int rowBase = blockIdx.x * RPB + wave * 16;
  if (rowBase >= M) return;              // wave-uniform exit; EXEC stays full for WMMA
  const int n0  = lane & 15;
  const int kbA = (lane < 16) ? 0 : 8;
  const int mo  = (lane < 16) ? 0 : 8;
  const int m   = rowBase + n0;
  const float sc = *msgScale;
  const float* rowS = S + (size_t)m * DHID;
  const float* rowM = msgs + (size_t)m * DHID;
  const float* rowF = rowS;
  if (FLIP) {
    int fm = (m < nv) ? (m + nv) : (m - nv);
    rowF = S + (size_t)fm * DHID;
  }
  constexpr int KT1 = FLIP ? 12 : 8;

  v8f acc[8] = {};
#pragma unroll
  for (int kt = 0; kt < KT1; ++kt) {
    const int seg = kt >> 2;
    const float* src = rowS;
    float s = 1.0f;
    if (seg == 1) { src = rowM; s = sc; }
    else if (FLIP && seg == 2) { src = rowF; }
    // preload all 8 B fragments for this k-tile, then burst 8 WMMAs
    v16bf bfr[8];
#pragma unroll
    for (int nt = 0; nt < 8; ++nt) bfr[nt] = load_b(W1p, kt, nt, lane);
    v16bf a = load_a_global(src, (kt & 3) * 32, kbA, s);
#pragma unroll
    for (int nt = 0; nt < 8; ++nt)
      acc[nt] = wmma_bf16(a, bfr[nt], acc[nt]);
  }

  // bias + relu -> LDS (bf16) ; C/D tile layout: elem(M = r+mo, N = nt*16+n0) = acc[nt][r]
  __bf16* Hl = Hb[wave];
#pragma unroll
  for (int nt = 0; nt < 8; ++nt) {
    float bb = b1[nt * 16 + n0];
#pragma unroll
    for (int r = 0; r < 8; ++r) {
      float h = acc[nt][r] + bb;
      Hl[(r + mo) * DHID + nt * 16 + n0] = (__bf16)(h > 0.0f ? h : 0.0f);
    }
  }
  asm volatile("s_wait_dscnt 0" ::: "memory");

  v8f acc2[8] = {};
#pragma unroll
  for (int kt = 0; kt < 4; ++kt) {
    v16bf bfr[8];
#pragma unroll
    for (int nt = 0; nt < 8; ++nt) bfr[nt] = load_b(W2p, kt, nt, lane);
    v16bf a = load_a_lds(Hl + n0 * DHID, kt * 32, kbA);
#pragma unroll
    for (int nt = 0; nt < 8; ++nt)
      acc2[nt] = wmma_bf16(a, bfr[nt], acc2[nt]);
  }

  float* Ol = Of[wave];
#pragma unroll
  for (int nt = 0; nt < 8; ++nt) {
    float bb = b2[nt * 16 + n0];
#pragma unroll
    for (int r = 0; r < 8; ++r)
      Ol[(r + mo) * DHID + nt * 16 + n0] = acc2[nt][r] + bb;
  }
  asm volatile("s_wait_dscnt 0" ::: "memory");

  // per-row normalize: 2 lanes per row, 64 cols each
  const int rr = lane >> 1;
  const int cb = (lane & 1) * 64;
  float s0 = 0.0f, s1 = 0.0f;
#pragma unroll 8
  for (int c = 0; c < 64; ++c) {
    float v = Ol[rr * DHID + cb + c];
    s0 += v; s1 += v * v;
  }
  s0 += __shfl_xor(s0, 1, 32);
  s1 += __shfl_xor(s1, 1, 32);
  float mean = s0 * (1.0f / 128.0f);
  float var  = (s1 - s0 * mean) * (1.0f / 127.0f);   // unbiased (ddof=1)
  var = var > 0.0f ? var : 0.0f;
  float inv = 1.0f / (sqrtf(var) + 1e-6f);
  const int grow = rowBase + rr;
  const float* oldRow = resid + (size_t)grow * DHID;
  float* outRow = out + (size_t)grow * DHID;
#pragma unroll 8
  for (int c = 0; c < 64; ++c) {
    int cc = cb + c;
    outRow[cc] = (Ol[rr * DHID + cc] - mean) * inv + oldRow[cc];
  }
}

// Final: V = [L[r], L[r+nv]] -> relu(V@VW1+Vb1) @ VW2 + Vb2 -> out[r]
__global__ void __launch_bounds__(128)
vout_kernel(const float* __restrict__ L, const __bf16* __restrict__ W1p,
            const float* __restrict__ b1, const float* __restrict__ W2,
            const float* __restrict__ b2, float* __restrict__ out, int nv) {
  __shared__ float Of[WPB][16 * DHID];
  const int wave = threadIdx.x >> 5;
  const int lane = threadIdx.x & 31;
  const int rowBase = blockIdx.x * RPB + wave * 16;
  if (rowBase >= nv) return;
  const int n0  = lane & 15;
  const int kbA = (lane < 16) ? 0 : 8;
  const int mo  = (lane < 16) ? 0 : 8;
  const int m   = rowBase + n0;
  const float* rowA = L + (size_t)m * DHID;
  const float* rowB = L + (size_t)(m + nv) * DHID;

  v8f acc[8] = {};
#pragma unroll
  for (int kt = 0; kt < 8; ++kt) {
    const float* src = (kt < 4) ? rowA : rowB;
    v16bf bfr[8];
#pragma unroll
    for (int nt = 0; nt < 8; ++nt) bfr[nt] = load_b(W1p, kt, nt, lane);
    v16bf a = load_a_global(src, (kt & 3) * 32, kbA, 1.0f);
#pragma unroll
    for (int nt = 0; nt < 8; ++nt)
      acc[nt] = wmma_bf16(a, bfr[nt], acc[nt]);
  }
  float* Ol = Of[wave];
#pragma unroll
  for (int nt = 0; nt < 8; ++nt) {
    float bb = b1[nt * 16 + n0];
#pragma unroll
    for (int r = 0; r < 8; ++r) {
      float h = acc[nt][r] + bb;
      Ol[(r + mo) * DHID + nt * 16 + n0] = h > 0.0f ? h : 0.0f;
    }
  }
  asm volatile("s_wait_dscnt 0" ::: "memory");
  const int rr = lane >> 1;
  const int cb = (lane & 1) * 64;
  float s = 0.0f;
#pragma unroll 8
  for (int c = 0; c < 64; ++c)
    s += Ol[rr * DHID + cb + c] * W2[cb + c];
  s += __shfl_xor(s, 1, 32);
  if ((lane & 1) == 0) out[rowBase + rr] = s + b2[0];
}

extern "C" void kernel_launch(void* const* d_in, const int* in_sizes, int n_in,
                              void* d_out, int out_size, void* d_ws, size_t ws_size,
                              hipStream_t stream) {
  const int NV = 100000, NC = 420000, T = 4;
  const int NL = 2 * NV;
  const int E  = in_sizes[3] / 2;
  const int* row = (const int*)d_in[3];      // clause index per edge
  const int* col = row + E;                  // literal index per edge
  const float* LinitS = (const float*)d_in[4];
  const float* CinitS = (const float*)d_in[5];
  const float* LCs = (const float*)d_in[6];
  const float* CLs = (const float*)d_in[7];
  const float* CW1 = (const float*)d_in[8];
  const float* Cb1 = (const float*)d_in[9];
  const float* CW2 = (const float*)d_in[10];
  const float* Cb2 = (const float*)d_in[11];
  const float* LW1 = (const float*)d_in[12];
  const float* Lb1 = (const float*)d_in[13];
  const float* LW2 = (const float*)d_in[14];
  const float* Lb2 = (const float*)d_in[15];
  const float* VW1 = (const float*)d_in[16];
  const float* Vb1 = (const float*)d_in[17];
  const float* VW2 = (const float*)d_in[18];
  const float* Vb2 = (const float*)d_in[19];
  float* outp = (float*)d_out;
  (void)n_in; (void)out_size; (void)ws_size;

  char* p = (char*)d_ws;
  auto take = [&](size_t bytes) {
    char* r = p;
    p += (bytes + 255) & ~(size_t)255;
    return r;
  };
  float* C0   = (float*)take((size_t)NC * DHID * 4);
  float* L0   = (float*)take((size_t)NL * DHID * 4);
  float* L1   = (float*)take((size_t)NL * DHID * 4);
  float* msgs = (float*)take((size_t)NC * DHID * 4);
  __bf16* CW1p = (__bf16*)take((size_t)T * 64 * 512 * 2);
  __bf16* CW2p = (__bf16*)take((size_t)T * 32 * 512 * 2);
  __bf16* LW1p = (__bf16*)take((size_t)T * 96 * 512 * 2);
  __bf16* LW2p = (__bf16*)take((size_t)T * 32 * 512 * 2);
  __bf16* VW1p = (__bf16*)take((size_t)64 * 512 * 2);

  // pack weights into WMMA B-fragment layout (bf16)
  for (int t = 0; t < T; ++t) {
    pack_kernel<<<8, 256, 0, stream>>>(CW1 + (size_t)t * 256 * DHID, CW1p + (size_t)t * 64 * 512, 8);
    pack_kernel<<<4, 256, 0, stream>>>(CW2 + (size_t)t * 128 * DHID, CW2p + (size_t)t * 32 * 512, 4);
    pack_kernel<<<12, 256, 0, stream>>>(LW1 + (size_t)t * 384 * DHID, LW1p + (size_t)t * 96 * 512, 12);
    pack_kernel<<<4, 256, 0, stream>>>(LW2 + (size_t)t * 128 * DHID, LW2p + (size_t)t * 32 * 512, 4);
  }
  pack_kernel<<<8, 256, 0, stream>>>(VW1, VW1p, 8);

  // init states
  fill_kernel<<<2048, 256, 0, stream>>>(L0, LinitS, (long)NL * DHID);
  fill_kernel<<<2048, 256, 0, stream>>>(C0, CinitS, (long)NC * DHID);

  float* Lc = L0;
  float* Ln = L1;
  const long sthreads = (long)E * 32;
  const int  sblocks  = (int)((sthreads + 255) / 256);

  for (int t = 0; t < T; ++t) {
    // LC_msgs = segsum(L[col] -> row)
    (void)hipMemsetAsync(msgs, 0, (size_t)NC * DHID * 4, stream);
    scatter_kernel<<<sblocks, 256, 0, stream>>>(Lc, msgs, col, row, E);
    mlp_round_kernel<false><<<(NC + RPB - 1) / RPB, 128, 0, stream>>>(
        C0, msgs, LCs,
        CW1p + (size_t)t * 64 * 512, Cb1 + (size_t)t * DHID,
        CW2p + (size_t)t * 32 * 512, Cb2 + (size_t)t * DHID,
        C0, C0, NC, 0);
    // CL_msgs = segsum(C[row] -> col)
    (void)hipMemsetAsync(msgs, 0, (size_t)NL * DHID * 4, stream);
    scatter_kernel<<<sblocks, 256, 0, stream>>>(C0, msgs, row, col, E);
    mlp_round_kernel<true><<<(NL + RPB - 1) / RPB, 128, 0, stream>>>(
        Lc, msgs, CLs,
        LW1p + (size_t)t * 96 * 512, Lb1 + (size_t)t * DHID,
        LW2p + (size_t)t * 32 * 512, Lb2 + (size_t)t * DHID,
        Lc, Ln, NL, NV);
    float* tmp = Lc; Lc = Ln; Ln = tmp;
  }

  vout_kernel<<<(NV + RPB - 1) / RPB, 128, 0, stream>>>(Lc, VW1p, Vb1, VW2, Vb2, outp, NV);
}
